// MengAIAdvanced_18442589569198
// MI455X (gfx1250) — compile-verified, matrix operand order checked by reading, not run on hardware
//
#include <hip/hip_runtime.h>

// ---------------------------------------------------------------------------
// MI455X (gfx1250) transformer forward.
// All GEMMs (QKV, attention scores/PV, WO, FFN, logits) run on
// v_wmma_f32_16x16x32_bf16 (wave32, fp32 accumulate). fp32 weights are
// converted to bf16 while staging into LDS; activations are kept as bf16
// shadow buffers. A-tiles stream into LDS with CDNA5 async copies
// (global_load_async_to_lds_b128 + s_wait_asynccnt). Attention is a
// streaming (flash) kernel with online softmax using wave32 shfl
// reductions that match the WMMA C-lane layout.
// ---------------------------------------------------------------------------

#define DEV __device__ __forceinline__

constexpr int S_  = 2048;
constexpr int D_  = 1024;
constexpr int H_  = 16;
constexpr int DK_ = 64;
constexpr int FF_ = 4096;
constexpr int V_  = 50000;
constexpr int L_  = 4;
constexpr float SCALE_ = 0.125f;   // 1/sqrt(64)

typedef __bf16 bf16_t;
typedef __attribute__((ext_vector_type(16))) bf16_t v16bf;
typedef __attribute__((ext_vector_type(8)))  float  v8f;

union Frag {
  v16bf v;
  unsigned short s[16];
};

DEV unsigned short f2bf(float f) {
  unsigned int u = __float_as_uint(f);
  u += 0x7fffu + ((u >> 16) & 1u);
  return (unsigned short)(u >> 16);
}

DEV v8f wmma_bf16(const Frag& a, const Frag& b, v8f c) {
  return __builtin_amdgcn_wmma_f32_16x16x32_bf16(
      false, a.v, false, b.v, (short)0, c, false, false);
}

// Load 32 bytes (16 bf16) into a fragment as two 16B vectors.
DEV void ld32(Frag& f, const unsigned short* p0, const unsigned short* p1) {
  *(uint4*)&f.s[0] = *(const uint4*)p0;
  *(uint4*)&f.s[8] = *(const uint4*)p1;
}

// LDS byte offset of a __shared__ pointer: flat LDS addresses carry the
// offset in addr[31:0] (ISA 10.2 aperture mapping), so truncation suffices.
DEV unsigned lds_lo(const void* p) { return (unsigned)(unsigned long long)p; }

// CDNA5 async copy: global -> LDS, 16B per lane, tracked by ASYNCcnt.
DEV void async_b128(unsigned lds, const void* g) {
  asm volatile("global_load_async_to_lds_b128 %0, %1, off"
               :: "v"(lds), "v"((unsigned long long)g)
               : "memory");
}
DEV void wait_async0() { asm volatile("s_wait_asynccnt 0x0" ::: "memory"); }

// ---------------------------------------------------------------------------
// GEMM: C[M,N] (f32) = A[M,K] (bf16) * B[K,N] (f32 -> bf16 in LDS) + bias
// 128x128 block tile, K-step 32, double-buffered LDS, 8 waves of 32x64.
// A-tile staged with async-to-LDS copies (double buffer = base + buf*8KB,
// loop-carried global pointers); B-tile loaded to regs (16 unconditional
// loads in flight; OOB columns are address-clamped, their garbage only feeds
// C columns that the guarded epilogue discards), converted, and written as
// packed 16B ds_store_b128 runs.
// ---------------------------------------------------------------------------
constexpr int TM = 128, TN = 128, TK = 32, BPAD = 8;

__global__ __launch_bounds__(256) void gemm_bf16_kernel(
    const unsigned short* __restrict__ A, const float* __restrict__ Bw,
    const float* __restrict__ bias, float* __restrict__ C,
    int M, int N, int K) {
  (void)M;
  __shared__ alignas(16) unsigned short sA[2][TM][TK];
  __shared__ alignas(16) unsigned short sB[2][TN][TK + BPAD];
  constexpr unsigned ABUF = TM * TK * 2;  // byte stride between sA buffers

  const int t = threadIdx.x;
  const int lane = t & 31, wave = t >> 5;
  const int wm = wave >> 1, wn = wave & 1;
  const int li = lane & 15, hi = lane >> 4;
  const int mBase = blockIdx.y * TM;
  const int nBase = blockIdx.x * TN;

  // --- per-thread staging geometry (constant across K-steps) ---
  // A: two 16B chunks per thread, loop-carried global pointers.
  const int arow = t >> 2;
  const int aseg = (t & 3) * 8;
  const unsigned short* aG0 = A + (size_t)(mBase + arow) * K + aseg;
  const unsigned short* aG1 = A + (size_t)(mBase + arow + 64) * K + aseg;
  const unsigned aL0 = lds_lo(&sA[0][arow][aseg]);
  const unsigned aL1 = lds_lo(&sA[0][arow + 64][aseg]);
  // B: one column (n), 16 consecutive k per thread; clamp OOB columns so all
  // loads are unconditional and in-bounds (their values are never stored).
  const int bn = t & 127;
  const int bkh = (t >> 7) * 16;
  const int gn = nBase + bn;
  const int gnc = (gn < N) ? gn : (N - 1);
  const float* bG = Bw + (size_t)bkh * N + gnc;

  auto stage = [&](int buf, int kBase) {
    const unsigned ao = (unsigned)buf * ABUF;
    async_b128(aL0 + ao, aG0);
    async_b128(aL1 + ao, aG1);
    float tv[16];
#pragma unroll
    for (int i = 0; i < 16; ++i)
      tv[i] = bG[(size_t)i * N];
    alignas(16) unsigned pk[8];
#pragma unroll
    for (int i = 0; i < 8; ++i)
      pk[i] = (unsigned)f2bf(tv[2 * i]) | ((unsigned)f2bf(tv[2 * i + 1]) << 16);
    *(uint4*)&sB[buf][bn][bkh] = *(const uint4*)&pk[0];
    *(uint4*)&sB[buf][bn][bkh + 8] = *(const uint4*)&pk[4];
    if (kBase + 2 * TK < K)
      __builtin_prefetch(bG + (size_t)2 * TK * N, 0, 1);
    // advance loop-carried staging pointers to the next K-tile
    aG0 += TK;
    aG1 += TK;
    bG += (size_t)TK * N;
  };

  v8f acc[2][4];
#pragma unroll
  for (int mi = 0; mi < 2; ++mi)
#pragma unroll
    for (int ni = 0; ni < 4; ++ni) acc[mi][ni] = (v8f)0.f;

  const int nk = K / TK;
  stage(0, 0);
  wait_async0();
  __syncthreads();

  for (int kt = 0; kt < nk; ++kt) {
    const int cur = kt & 1;
    if (kt + 1 < nk) stage(cur ^ 1, (kt + 1) * TK);
    Frag aF[2], bF[4];
#pragma unroll
    for (int mi = 0; mi < 2; ++mi) {
      const unsigned short* p = &sA[cur][wm * 32 + mi * 16 + li][hi * 8];
      ld32(aF[mi], p, p + 16);
    }
#pragma unroll
    for (int ni = 0; ni < 4; ++ni) {
      const unsigned short* p = &sB[cur][wn * 64 + ni * 16 + li][hi * 8];
      ld32(bF[ni], p, p + 16);
    }
#pragma unroll
    for (int mi = 0; mi < 2; ++mi)
#pragma unroll
      for (int ni = 0; ni < 4; ++ni)
        acc[mi][ni] = wmma_bf16(aF[mi], bF[ni], acc[mi][ni]);
    wait_async0();
    __syncthreads();
  }

#pragma unroll
  for (int mi = 0; mi < 2; ++mi)
#pragma unroll
    for (int ni = 0; ni < 4; ++ni) {
      const int col = nBase + wn * 64 + ni * 16 + li;
      if (col < N) {
        const float bv = bias ? bias[col] : 0.f;
#pragma unroll
        for (int r = 0; r < 8; ++r) {
          const int row = mBase + wm * 32 + mi * 16 + hi * 8 + r;
          C[(size_t)row * N + col] = acc[mi][ni][r] + bv;
        }
      }
    }
}

// ---------------------------------------------------------------------------
// Flash attention: block = (128 queries, 1 head), 8 waves x 16 queries each.
// Streams key blocks of 32, online softmax, ctx written as bf16 (S,D).
// qb/kb: bf16 (S,D) post-RoPE.  vt: bf16 (H*DK, S) = V transposed.
// ---------------------------------------------------------------------------
__global__ __launch_bounds__(256) void attention_kernel(
    const unsigned short* __restrict__ qb, const unsigned short* __restrict__ kbuf,
    const unsigned short* __restrict__ vt, unsigned short* __restrict__ ctx) {
  __shared__ alignas(16) unsigned short pLds[8][16][32];

  const int t = threadIdx.x, lane = t & 31, w = t >> 5;
  const int li = lane & 15, hi = lane >> 4;
  const int h = blockIdx.y;
  const int qBlock = blockIdx.x;
  const int qTile = qBlock * 128 + w * 16;
  const int fBase = h * DK_;

  // Q fragments: 16 rows x 64 features -> two 16x32 A-frags.
  Frag aQ[2];
  {
    const unsigned short* qrow = qb + (size_t)(qTile + li) * D_ + fBase;
#pragma unroll
    for (int f = 0; f < 2; ++f)
      ld32(aQ[f], qrow + f * 32 + hi * 8, qrow + f * 32 + 16 + hi * 8);
  }

  v8f o[4];
#pragma unroll
  for (int ni = 0; ni < 4; ++ni) o[ni] = (v8f)0.f;
  float mrow[8], lrow[8];
#pragma unroll
  for (int r = 0; r < 8; ++r) { mrow[r] = -3.0e38f; lrow[r] = 0.f; }

  const int nkb = (qBlock + 1) * 4;  // 32-key blocks covering causal span
  for (int kbi = 0; kbi < nkb; ++kbi) {
    const int key0 = kbi * 32;

    // scores: 16q x 32keys as two 16x16 WMMA accumulators
    v8f sc[2];
    sc[0] = (v8f)0.f;
    sc[1] = (v8f)0.f;
#pragma unroll
    for (int tt = 0; tt < 2; ++tt) {
      const unsigned short* krow =
          kbuf + (size_t)(key0 + tt * 16 + li) * D_ + fBase;
#pragma unroll
      for (int f = 0; f < 2; ++f) {
        Frag bK;
        ld32(bK, krow + f * 32 + hi * 8, krow + f * 32 + 16 + hi * 8);
        sc[tt] = wmma_bf16(aQ[f], bK, sc[tt]);
      }
    }

    // masked online softmax; rows live in regs, cols in lanes (width-16 shfl)
#pragma unroll
    for (int r = 0; r < 8; ++r) {
      const int qrow = qTile + hi * 8 + r;
      float s0 = sc[0][r] * SCALE_;
      float s1 = sc[1][r] * SCALE_;
      if (key0 + li > qrow) s0 = -1e9f;
      if (key0 + 16 + li > qrow) s1 = -1e9f;
      float cm = fmaxf(s0, s1);
#pragma unroll
      for (int mm = 1; mm < 16; mm <<= 1) cm = fmaxf(cm, __shfl_xor(cm, mm, 16));
      const float mnew = fmaxf(mrow[r], cm);
      const float scalef = __expf(mrow[r] - mnew);
      const float e0 = __expf(s0 - mnew);
      const float e1 = __expf(s1 - mnew);
      float psum = e0 + e1;
#pragma unroll
      for (int mm = 1; mm < 16; mm <<= 1) psum += __shfl_xor(psum, mm, 16);
      lrow[r] = lrow[r] * scalef + psum;
      mrow[r] = mnew;
#pragma unroll
      for (int ni = 0; ni < 4; ++ni) o[ni][r] *= scalef;
      pLds[w][hi * 8 + r][li] = f2bf(e0);
      pLds[w][hi * 8 + r][16 + li] = f2bf(e1);
    }
    // per-wave LDS transpose of P -> A-fragment layout
    asm volatile("s_wait_dscnt 0x0" ::: "memory");
    __builtin_amdgcn_wave_barrier();
    Frag aP;
    {
      const unsigned short* pr = &pLds[w][li][hi * 8];
      ld32(aP, pr, pr + 16);
    }
    asm volatile("" ::: "memory");

    // O += P (16x32) * V (32x64), V pre-transposed so B-frags are contiguous
#pragma unroll
    for (int ni = 0; ni < 4; ++ni) {
      const unsigned short* vrow =
          vt + (size_t)(fBase + ni * 16 + li) * S_ + key0;
      Frag bV;
      ld32(bV, vrow + hi * 8, vrow + 16 + hi * 8);
      o[ni] = wmma_bf16(aP, bV, o[ni]);
    }
  }

#pragma unroll
  for (int ni = 0; ni < 4; ++ni)
#pragma unroll
    for (int r = 0; r < 8; ++r) {
      const int row = qTile + hi * 8 + r;
      const float val = o[ni][r] / lrow[r];
      ctx[(size_t)row * D_ + fBase + ni * 16 + li] = f2bf(val);
    }
}

// ---------------------------------------------------------------------------
// Elementwise kernels
// ---------------------------------------------------------------------------
__global__ void embed_kernel(const int* __restrict__ ids,
                             const float* __restrict__ tok,
                             const float* __restrict__ pos,
                             float* __restrict__ x,
                             unsigned short* __restrict__ xb) {
  const int s = blockIdx.x;
  const int id = ids[s];
  for (int d = threadIdx.x; d < D_; d += blockDim.x) {
    const float v = tok[(size_t)id * D_ + d] + pos[(size_t)s * D_ + d];
    x[(size_t)s * D_ + d] = v;
    xb[(size_t)s * D_ + d] = f2bf(v);
  }
}

__global__ void rope_bf16_kernel(const float* __restrict__ in,
                                 unsigned short* __restrict__ out) {
  const size_t idx = (size_t)blockIdx.x * 256 + threadIdx.x;  // S*D
  const int d = (int)(idx & (D_ - 1));
  const int s = (int)(idx >> 10);
  const int i = d & (DK_ - 1);
  const int io = i & (DK_ / 2 - 1);
  const float freq = __powf(10000.f, -(float)(2 * io) * (1.f / DK_));
  const float ang = (float)s * freq;
  float sn, c;
  __sincosf(ang, &sn, &c);
  const float xv = in[idx];
  const float xr = (i < DK_ / 2) ? -in[idx + DK_ / 2] : in[idx - DK_ / 2];
  out[idx] = f2bf(xv * c + xr * sn);
}

__global__ void vtrans_bf16_kernel(const float* __restrict__ in,
                                   unsigned short* __restrict__ out) {
  const size_t idx = (size_t)blockIdx.x * 256 + threadIdx.x;  // S*D
  const int s = (int)(idx & (S_ - 1));
  const int hd = (int)(idx >> 11);
  out[(size_t)hd * S_ + s] = f2bf(in[(size_t)s * D_ + hd]);
}

__global__ void gelu_bf16_kernel(const float* __restrict__ in,
                                 unsigned short* __restrict__ out) {
  const size_t idx = (size_t)blockIdx.x * 256 + threadIdx.x;  // S*FF
  const float x = in[idx];
  out[idx] = f2bf(0.5f * x * (1.f + erff(x * 0.70710678118f)));
}

DEV float block_reduce_sum(float v, float* sm) {
#pragma unroll
  for (int m = 1; m < 32; m <<= 1) v += __shfl_xor(v, m, 32);
  const int w = threadIdx.x >> 5;
  __syncthreads();
  if ((threadIdx.x & 31) == 0) sm[w] = v;
  __syncthreads();
  float r = sm[0];
#pragma unroll
  for (int i = 1; i < 8; ++i) r += sm[i];
  return r;
}

__global__ __launch_bounds__(256) void add_ln_kernel(
    const float* __restrict__ xin, const float* __restrict__ yin,
    const float* __restrict__ w, const float* __restrict__ b,
    float* __restrict__ xout, unsigned short* __restrict__ xbf) {
  __shared__ float sm[8];
  const int s = blockIdx.x;
  const int t = threadIdx.x;
  const size_t base = (size_t)s * D_;
  float v[4];
  float sum = 0.f;
#pragma unroll
  for (int i = 0; i < 4; ++i) {
    const int d = t + i * 256;
    v[i] = xin[base + d] + yin[base + d];
    sum += v[i];
  }
  sum = block_reduce_sum(sum, sm);
  const float mu = sum * (1.f / D_);
  float var = 0.f;
#pragma unroll
  for (int i = 0; i < 4; ++i) { const float c = v[i] - mu; var += c * c; }
  var = block_reduce_sum(var, sm) * (1.f / D_);
  const float rs = rsqrtf(var + 1e-5f);
#pragma unroll
  for (int i = 0; i < 4; ++i) {
    const int d = t + i * 256;
    const float o = (v[i] - mu) * rs * w[d] + b[d];
    xout[base + d] = o;
    xbf[base + d] = f2bf(o);
  }
}

// ---------------------------------------------------------------------------
// Host orchestration
// ---------------------------------------------------------------------------
extern "C" void kernel_launch(void* const* d_in, const int* in_sizes, int n_in,
                              void* d_out, int out_size, void* d_ws,
                              size_t ws_size, hipStream_t stream) {
  (void)in_sizes; (void)n_in; (void)out_size; (void)ws_size;

  const int*   ids  = (const int*)d_in[0];
  const float* tok  = (const float*)d_in[1];
  const float* pos  = (const float*)d_in[2];
  const float* wq   = (const float*)d_in[3];
  const float* wk   = (const float*)d_in[4];
  const float* wv   = (const float*)d_in[5];
  const float* wo_w = (const float*)d_in[6];
  const float* wo_b = (const float*)d_in[7];
  const float* ln1w = (const float*)d_in[8];
  const float* ln1b = (const float*)d_in[9];
  const float* ln2w = (const float*)d_in[10];
  const float* ln2b = (const float*)d_in[11];
  const float* ff1w = (const float*)d_in[12];
  const float* ff1b = (const float*)d_in[13];
  const float* ff2w = (const float*)d_in[14];
  const float* ff2b = (const float*)d_in[15];
  const float* outw = (const float*)d_in[16];
  const float* outb = (const float*)d_in[17];

  char* ws = (char*)d_ws;
  size_t off = 0;
  auto carve = [&](size_t bytes) {
    char* p = ws + off;
    off = (off + bytes + 255) & ~(size_t)255;
    return p;
  };
  float*          x   = (float*)carve((size_t)S_ * D_ * 4);
  unsigned short* xb  = (unsigned short*)carve((size_t)S_ * D_ * 2);
  float*          scr = (float*)carve((size_t)S_ * FF_ * 4);
  unsigned short* qb  = (unsigned short*)carve((size_t)S_ * D_ * 2);
  unsigned short* kb  = (unsigned short*)carve((size_t)S_ * D_ * 2);
  unsigned short* vt  = (unsigned short*)carve((size_t)S_ * D_ * 2);
  unsigned short* cb  = (unsigned short*)carve((size_t)S_ * D_ * 2);
  unsigned short* hb  = (unsigned short*)carve((size_t)S_ * FF_ * 2);

  const dim3 blk(256);
  const int nSD = S_ * D_ / 256;
  const int nSF = S_ * FF_ / 256;

  embed_kernel<<<S_, blk, 0, stream>>>(ids, tok, pos, x, xb);

  auto gemm = [&](const unsigned short* A, const float* B, const float* bias,
                  float* C, int N, int K) {
    dim3 grid((N + TN - 1) / TN, S_ / TM);
    gemm_bf16_kernel<<<grid, blk, 0, stream>>>(A, B, bias, C, S_, N, K);
  };

  for (int l = 0; l < L_; ++l) {
    const size_t dd = (size_t)D_ * D_;
    gemm(xb, wq + l * dd, nullptr, scr, D_, D_);
    rope_bf16_kernel<<<nSD, blk, 0, stream>>>(scr, qb);
    gemm(xb, wk + l * dd, nullptr, scr, D_, D_);
    rope_bf16_kernel<<<nSD, blk, 0, stream>>>(scr, kb);
    gemm(xb, wv + l * dd, nullptr, scr, D_, D_);
    vtrans_bf16_kernel<<<nSD, blk, 0, stream>>>(scr, vt);

    attention_kernel<<<dim3(S_ / 128, H_), blk, 0, stream>>>(qb, kb, vt, cb);

    gemm(cb, wo_w + l * dd, wo_b + (size_t)l * D_, scr, D_, D_);
    add_ln_kernel<<<S_, blk, 0, stream>>>(x, scr, ln1w + (size_t)l * D_,
                                          ln1b + (size_t)l * D_, x, xb);

    gemm(xb, ff1w + (size_t)l * D_ * FF_, ff1b + (size_t)l * FF_, scr, FF_, D_);
    gelu_bf16_kernel<<<nSF, blk, 0, stream>>>(scr, hb);
    gemm(hb, ff2w + (size_t)l * FF_ * D_, ff2b + (size_t)l * D_, scr, D_, FF_);
    add_ln_kernel<<<S_, blk, 0, stream>>>(x, scr, ln2w + (size_t)l * D_,
                                          ln2b + (size_t)l * D_, x, xb);
  }

  gemm(xb, outw, outb, (float*)d_out, V_, D_);
}
